// ReferenceMoE_52578989638366
// MI455X (gfx1250) — compile-verified, matrix operand order checked by reading, not run on hardware
//
#include <hip/hip_runtime.h>

// ------------------------------------------------------------------
// MoE (top-2) for MI455X / gfx1250.
// f32 inputs -> bf16 WMMA (v_wmma_f32_16x16x32_bf16) with f32 accum.
// Sparse: only the 2048 assigned (token,expert) rows are computed.
// Deterministic (ordered compaction, no atomics) for graph replay.
// Double-buffered LDS pipeline; cheap packed bf16 staging
// (v_add + v_perm round-half-up; native v_cvt_pk_bf16_f32 if the
// toolchain exposes it); global_prefetch for L2 warming; rcp silu.
// ------------------------------------------------------------------

constexpr int NUM_EXPERTS = 8;
constexpr int NUM_ROWS    = 1024;
constexpr int HIDDEN      = 2048;
constexpr int INTER       = 1024;
constexpr int MAXSLOTS    = NUM_ROWS * 2;   // top-2 -> 2048 assigned rows

// GEMM tiling: 256 threads = 8 waves (wave32). Block tile 64x64, K-step 32.
constexpr int BM  = 64;
constexpr int BN  = 64;
constexpr int BK  = 32;
constexpr int BKP = BK + 8;                 // padded LDS stride (80 B rows)

typedef __bf16 bf16_t;
typedef __attribute__((ext_vector_type(16))) bf16_t v16bf;
typedef __attribute__((ext_vector_type(8)))  float  v8f;

// f32 -> packed bf16x2 (lo in [15:0])
#if __has_builtin(__builtin_amdgcn_cvt_pk_bf16_f32)
typedef __attribute__((ext_vector_type(2))) bf16_t v2bf;
__device__ __forceinline__ unsigned pack_bf16x2(float lo, float hi) {
  union { v2bf v; unsigned u; } t;
  t.v = __builtin_amdgcn_cvt_pk_bf16_f32(lo, hi);   // one v_cvt_pk_bf16_f32
  return t.u;
}
__device__ __forceinline__ unsigned short cvt_bf16(float f) {
  return (unsigned short)(pack_bf16x2(f, 0.f) & 0xffffu);
}
#else
// round-half-up: 2x v_add_nc_u32 + 1x v_perm_b32 per pair (no tie-to-even
// LSB logic -- that was costing v_bfe/v_and/v_mov_b16 chains per element)
__device__ __forceinline__ unsigned pack_bf16x2(float lo, float hi) {
  unsigned a = __float_as_uint(lo) + 0x8000u;
  unsigned b = __float_as_uint(hi) + 0x8000u;
  return __builtin_amdgcn_perm(b, a, 0x07060302u);  // { b[31:16], a[31:16] }
}
__device__ __forceinline__ unsigned short cvt_bf16(float f) {
  return (unsigned short)((__float_as_uint(f) + 0x8000u) >> 16);
}
#endif

// 16 bf16 fragment from LDS as two 16B loads (rows are 16B aligned, 80B stride)
__device__ __forceinline__ v16bf load_frag16(const unsigned short* p) {
  union { uint4 u[2]; v16bf v; } t;
  t.u[0] = *(const uint4*)p;
  t.u[1] = *(const uint4*)(p + 8);
  return t.v;
}

// ------------------------------------------------------------------
// Kernel 1: router softmax + top-2 + renorm + deterministic compaction
// ------------------------------------------------------------------
__global__ __launch_bounds__(1024)
void moe_router_kernel(const float* __restrict__ logits,
                       float* __restrict__ slotw,     // [NUM_ROWS][2]
                       int*   __restrict__ rowToken,  // [MAXSLOTS] packed token*2+slot
                       int*   __restrict__ meta)      // [0..7]=base, [8..15]=cnt
{
  __shared__ unsigned char sIdx[NUM_ROWS][2];
  __shared__ int sBase[NUM_EXPERTS + 1];
  __shared__ int sCnt[NUM_EXPERTS];

  const int n = threadIdx.x;
  float p[NUM_EXPERTS];
  float mx = -1e30f;
  #pragma unroll
  for (int e = 0; e < NUM_EXPERTS; ++e) { p[e] = logits[n * NUM_EXPERTS + e]; mx = fmaxf(mx, p[e]); }
  #pragma unroll
  for (int e = 0; e < NUM_EXPERTS; ++e) p[e] = __expf(p[e] - mx);
  // top-2 on exp values (== top-2 on softmax); ties -> lower index (matches lax.top_k)
  int i0 = 0; float v0 = p[0];
  #pragma unroll
  for (int e = 1; e < NUM_EXPERTS; ++e) if (p[e] > v0) { v0 = p[e]; i0 = e; }
  int i1 = (i0 == 0) ? 1 : 0; float v1 = p[i1];
  #pragma unroll
  for (int e = 0; e < NUM_EXPERTS; ++e) if (e != i0 && p[e] > v1) { v1 = p[e]; i1 = e; }
  const float inv = __builtin_amdgcn_rcpf(v0 + v1);   // softmax denom cancels in renorm
  slotw[n * 2 + 0] = v0 * inv;
  slotw[n * 2 + 1] = v1 * inv;
  sIdx[n][0] = (unsigned char)i0;
  sIdx[n][1] = (unsigned char)i1;
  __syncthreads();

  if (threadIdx.x < NUM_EXPERTS) {          // per-expert histogram (ordered, deterministic)
    const int e = threadIdx.x; int c = 0;
    for (int t = 0; t < NUM_ROWS; ++t) c += (sIdx[t][0] == e) + (sIdx[t][1] == e);
    sCnt[e] = c;
  }
  __syncthreads();
  if (threadIdx.x == 0) {                   // exclusive prefix over 8 experts
    int acc = 0;
    for (int e = 0; e < NUM_EXPERTS; ++e) { sBase[e] = acc; acc += sCnt[e]; }
    sBase[NUM_EXPERTS] = acc;
    for (int e = 0; e < NUM_EXPERTS; ++e) { meta[e] = sBase[e]; meta[NUM_EXPERTS + e] = sCnt[e]; }
  }
  __syncthreads();
  if (threadIdx.x < NUM_EXPERTS) {          // ordered assignment -> deterministic lists
    const int e = threadIdx.x; int c = sBase[e];
    for (int t = 0; t < NUM_ROWS; ++t) {
      if (sIdx[t][0] == e) rowToken[c++] = t * 2 + 0;
      if (sIdx[t][1] == e) rowToken[c++] = t * 2 + 1;
    }
  }
}

// ------------------------------------------------------------------
// Kernel 2: gathered GEMM x@fc1 / x@fc3 fused with SwiGLU -> bf16 act
// ------------------------------------------------------------------
__global__ __launch_bounds__(256)
void moe_gemm1_swiglu(const float* __restrict__ x,
                      const float* __restrict__ fc1,
                      const float* __restrict__ fc3,
                      const int*   __restrict__ rowToken,
                      const int*   __restrict__ meta,
                      unsigned short* __restrict__ act)   // [MAXSLOTS][INTER] bf16
{
  const int e    = blockIdx.z;
  const int base = meta[e];
  const int cnt  = meta[NUM_EXPERTS + e];
  const int m0   = blockIdx.y * BM;
  if (m0 >= cnt) return;                    // uniform early exit
  const int n0   = blockIdx.x * BN;

  __shared__ alignas(16) unsigned short As [2][BM][BKP];  // row-major [m][k]
  __shared__ alignas(16) unsigned short B1s[2][BN][BKP];  // transposed [n][k]
  __shared__ alignas(16) unsigned short B3s[2][BN][BKP];
  __shared__ int sTok[BM];

  const int tid = threadIdx.x;
  if (tid < BM) {
    const int m = m0 + tid;
    sTok[tid] = (m < cnt) ? (rowToken[base + m] >> 1) : 0;
  }
  __syncthreads();

  const int lane = tid & 31;
  const int wave = tid >> 5;
  const int wm   = wave & 3;                // M offset = 16*wm
  const int wn   = wave >> 2;               // N offset = 32*wn
  const int lr   = lane & 15;               // fragment row/col within 16
  const int lkh  = (lane >> 4) << 4;        // K half: 0 or 16 (contiguous k = 16h + s, same order for A and B)
  const int lm8  = (lane >> 4) << 3;        // C layout: lanes 16-31 -> M += 8

  v8f acc1[2], acc3[2];
  { v8f z = {0.f,0.f,0.f,0.f,0.f,0.f,0.f,0.f}; acc1[0]=z; acc1[1]=z; acc3[0]=z; acc3[1]=z; }

  const size_t fcBase = (size_t)e * HIDDEN * INTER;
  const int arow = tid >> 2, acol = (tid & 3) << 3;   // A staging: 64 rows x 8 k
  const int bkk  = (tid >> 4) << 1;                   // B staging: even k (2 rows)
  const int bn4  = (tid & 15) << 2;                   // 4 consecutive n

  float4 a0, a1, r0, r1, q0, q1;            // in-flight staging registers

  auto gload = [&](int k0) {
    const float4* s = (const float4*)(x + (size_t)sTok[arow] * HIDDEN + k0 + acol);
    a0 = s[0]; a1 = s[1];
    const float* p1 = fc1 + fcBase + (size_t)(k0 + bkk) * INTER + n0 + bn4;
    const float* p3 = fc3 + fcBase + (size_t)(k0 + bkk) * INTER + n0 + bn4;
    r0 = *(const float4*)p1; r1 = *(const float4*)(p1 + INTER);
    q0 = *(const float4*)p3; q1 = *(const float4*)(p3 + INTER);
  };
  auto lstore = [&](int buf) {
    uint4 aw;
    aw.x = pack_bf16x2(a0.x, a0.y); aw.y = pack_bf16x2(a0.z, a0.w);
    aw.z = pack_bf16x2(a1.x, a1.y); aw.w = pack_bf16x2(a1.z, a1.w);
    *(uint4*)&As[buf][arow][acol] = aw;                       // one ds_store_b128
    *(unsigned*)&B1s[buf][bn4 + 0][bkk] = pack_bf16x2(r0.x, r1.x);
    *(unsigned*)&B1s[buf][bn4 + 1][bkk] = pack_bf16x2(r0.y, r1.y);
    *(unsigned*)&B1s[buf][bn4 + 2][bkk] = pack_bf16x2(r0.z, r1.z);
    *(unsigned*)&B1s[buf][bn4 + 3][bkk] = pack_bf16x2(r0.w, r1.w);
    *(unsigned*)&B3s[buf][bn4 + 0][bkk] = pack_bf16x2(q0.x, q1.x);
    *(unsigned*)&B3s[buf][bn4 + 1][bkk] = pack_bf16x2(q0.y, q1.y);
    *(unsigned*)&B3s[buf][bn4 + 2][bkk] = pack_bf16x2(q0.z, q1.z);
    *(unsigned*)&B3s[buf][bn4 + 3][bkk] = pack_bf16x2(q0.w, q1.w);
  };

  constexpr int NK = HIDDEN / BK;
  gload(0);
  lstore(0);
  __syncthreads();

  for (int kt = 0; kt < NK; ++kt) {
    const int  cur     = kt & 1;
    const bool hasNext = (kt + 1) < NK;
    if (hasNext) {
      gload((kt + 1) * BK);                 // global loads overlap WMMAs below
      if (kt + 2 < NK) {                    // warm L2 one tile further ahead
        __builtin_prefetch(fc1 + fcBase + (size_t)((kt + 2) * BK + bkk) * INTER + n0 + bn4, 0, 1);
        __builtin_prefetch(fc3 + fcBase + (size_t)((kt + 2) * BK + bkk) * INTER + n0 + bn4, 0, 1);
      }
    }
    const v16bf a = load_frag16(&As[cur][wm * 16 + lr][lkh]);
    #pragma unroll
    for (int s = 0; s < 2; ++s) {
      const v16bf b1 = load_frag16(&B1s[cur][wn * 32 + s * 16 + lr][lkh]);
      const v16bf b3 = load_frag16(&B3s[cur][wn * 32 + s * 16 + lr][lkh]);
      acc1[s] = __builtin_amdgcn_wmma_f32_16x16x32_bf16(false, a, false, b1, (short)0, acc1[s], false, false);
      acc3[s] = __builtin_amdgcn_wmma_f32_16x16x32_bf16(false, a, false, b3, (short)0, acc3[s], false, false);
    }
    if (hasNext) lstore(cur ^ 1);
    __syncthreads();                        // single barrier per K-step (double buffer)
  }

  #pragma unroll
  for (int s = 0; s < 2; ++s) {
    #pragma unroll
    for (int v = 0; v < 8; ++v) {
      const int ml = wm * 16 + v + lm8;      // C/D layout: VGPR v, lane halves -> M
      const int m  = m0 + ml;
      if (m < cnt) {
        const float h1 = acc1[s][v];
        const float h3 = acc3[s][v];
        const float sw = h1 * __builtin_amdgcn_rcpf(1.f + __expf(-h1));   // silu via v_rcp_f32
        const int col = n0 + wn * 32 + s * 16 + lr;
        act[(size_t)(base + m) * INTER + col] = cvt_bf16(sw * h3);
      }
    }
  }
}

// ------------------------------------------------------------------
// Kernel 3: act @ fc2 -> scatter f32 rows into per-(token,slot) buffer
// ------------------------------------------------------------------
__global__ __launch_bounds__(256)
void moe_gemm2_scatter(const unsigned short* __restrict__ act,
                       const float* __restrict__ fc2,
                       const int*   __restrict__ rowToken,
                       const int*   __restrict__ meta,
                       float* __restrict__ h2)          // [MAXSLOTS][HIDDEN] f32
{
  const int e    = blockIdx.z;
  const int base = meta[e];
  const int cnt  = meta[NUM_EXPERTS + e];
  const int m0   = blockIdx.y * BM;
  if (m0 >= cnt) return;
  const int n0   = blockIdx.x * BN;

  __shared__ alignas(16) unsigned short As[2][BM][BKP];
  __shared__ alignas(16) unsigned short Bs[2][BN][BKP];   // transposed [n][k]
  __shared__ int sDst[BM];

  const int tid = threadIdx.x;
  if (tid < BM) {
    const int m = m0 + tid;
    sDst[tid] = (m < cnt) ? rowToken[base + m] : -1;
  }
  __syncthreads();

  const int lane = tid & 31;
  const int wave = tid >> 5;
  const int wm = wave & 3, wn = wave >> 2;
  const int lr  = lane & 15;
  const int lkh = (lane >> 4) << 4;
  const int lm8 = (lane >> 4) << 3;

  v8f acc[2];
  { v8f z = {0.f,0.f,0.f,0.f,0.f,0.f,0.f,0.f}; acc[0]=z; acc[1]=z; }

  const size_t fcBase = (size_t)e * INTER * HIDDEN;
  const int arow = tid >> 2, acol = (tid & 3) << 3;
  const int bkk  = (tid >> 4) << 1;
  const int bn4  = (tid & 15) << 2;
  int srcRow = base + m0 + arow;
  if (srcRow >= MAXSLOTS) srcRow = 0;       // padded rows: loads harmless, stores guarded

  uint4 aw; float4 r0, r1;
  auto gload = [&](int k0) {
    aw = *(const uint4*)(act + (size_t)srcRow * INTER + k0 + acol);   // already bf16
    const float* p = fc2 + fcBase + (size_t)(k0 + bkk) * HIDDEN + n0 + bn4;
    r0 = *(const float4*)p; r1 = *(const float4*)(p + HIDDEN);
  };
  auto lstore = [&](int buf) {
    *(uint4*)&As[buf][arow][acol] = aw;
    *(unsigned*)&Bs[buf][bn4 + 0][bkk] = pack_bf16x2(r0.x, r1.x);
    *(unsigned*)&Bs[buf][bn4 + 1][bkk] = pack_bf16x2(r0.y, r1.y);
    *(unsigned*)&Bs[buf][bn4 + 2][bkk] = pack_bf16x2(r0.z, r1.z);
    *(unsigned*)&Bs[buf][bn4 + 3][bkk] = pack_bf16x2(r0.w, r1.w);
  };

  constexpr int NK = INTER / BK;
  gload(0);
  lstore(0);
  __syncthreads();

  for (int kt = 0; kt < NK; ++kt) {
    const int  cur     = kt & 1;
    const bool hasNext = (kt + 1) < NK;
    if (hasNext) {
      gload((kt + 1) * BK);
      if (kt + 2 < NK)
        __builtin_prefetch(fc2 + fcBase + (size_t)((kt + 2) * BK + bkk) * HIDDEN + n0 + bn4, 0, 1);
    }
    const v16bf a = load_frag16(&As[cur][wm * 16 + lr][lkh]);
    #pragma unroll
    for (int s = 0; s < 2; ++s) {
      const v16bf b = load_frag16(&Bs[cur][wn * 32 + s * 16 + lr][lkh]);
      acc[s] = __builtin_amdgcn_wmma_f32_16x16x32_bf16(false, a, false, b, (short)0, acc[s], false, false);
    }
    if (hasNext) lstore(cur ^ 1);
    __syncthreads();
  }

  #pragma unroll
  for (int s = 0; s < 2; ++s) {
    #pragma unroll
    for (int v = 0; v < 8; ++v) {
      const int ml  = wm * 16 + v + lm8;
      const int dst = sDst[ml];
      if (dst >= 0) {
        const int col = n0 + wn * 32 + s * 16 + lr;
        h2[(size_t)dst * HIDDEN + col] = acc[s][v];
      }
    }
  }
}

// ------------------------------------------------------------------
// Kernel 4: out[n] = w0*h2[n,slot0] + w1*h2[n,slot1]  (float4)
// ------------------------------------------------------------------
__global__ __launch_bounds__(256)
void moe_combine_kernel(const float* __restrict__ h2,
                        const float* __restrict__ slotw,
                        float* __restrict__ out)
{
  const int gid = blockIdx.x * 256 + threadIdx.x;      // one float4 each
  const int n = gid >> 9;                              // HIDDEN/4 = 512 per row
  const int c = (gid & 511) << 2;
  const float w0 = slotw[n * 2 + 0];
  const float w1 = slotw[n * 2 + 1];
  const float4 a = *(const float4*)&h2[(size_t)(n * 2 + 0) * HIDDEN + c];
  const float4 b = *(const float4*)&h2[(size_t)(n * 2 + 1) * HIDDEN + c];
  float4 o;
  o.x = w0 * a.x + w1 * b.x;
  o.y = w0 * a.y + w1 * b.y;
  o.z = w0 * a.z + w1 * b.z;
  o.w = w0 * a.w + w1 * b.w;
  *(float4*)&out[(size_t)n * HIDDEN + c] = o;
}

// ------------------------------------------------------------------
// Host-side launcher (harness entry point)
// ------------------------------------------------------------------
extern "C" void kernel_launch(void* const* d_in, const int* in_sizes, int n_in,
                              void* d_out, int out_size, void* d_ws, size_t ws_size,
                              hipStream_t stream) {
  (void)in_sizes; (void)n_in; (void)out_size; (void)ws_size;
  const float* x      = (const float*)d_in[0];
  const float* router = (const float*)d_in[1];
  const float* fc1    = (const float*)d_in[2];
  const float* fc2    = (const float*)d_in[3];
  const float* fc3    = (const float*)d_in[4];
  float* out = (float*)d_out;

  // Workspace layout (bytes), total ~20.0 MB:
  //   [0,       8192)  float slotw[2048]
  //   [8192,   16384)  int   rowToken[2048]
  //   [16384,  16640)  int   meta[16] (base[8], cnt[8]) + pad
  //   [16640, +4 MB )  bf16  act[2048][1024]
  //   [+4MB,  +16 MB)  f32   h2[2048][2048]
  char* ws = (char*)d_ws;
  float*          slotw    = (float*)(ws + 0);
  int*            rowToken = (int*)(ws + 8192);
  int*            meta     = (int*)(ws + 16384);
  unsigned short* act      = (unsigned short*)(ws + 16640);
  float*          h2       = (float*)(ws + 16640 + (size_t)MAXSLOTS * INTER * 2);

  moe_router_kernel<<<1, 1024, 0, stream>>>(router, slotw, rowToken, meta);

  dim3 g1(INTER / BN, NUM_ROWS / BM, NUM_EXPERTS);     // worst-case M tiles; blocks early-exit
  moe_gemm1_swiglu<<<g1, 256, 0, stream>>>(x, fc1, fc3, rowToken, meta, act);

  dim3 g2(HIDDEN / BN, NUM_ROWS / BM, NUM_EXPERTS);
  moe_gemm2_scatter<<<g2, 256, 0, stream>>>(act, fc2, rowToken, meta, h2);

  moe_combine_kernel<<<(NUM_ROWS * HIDDEN / 4) / 256, 256, 0, stream>>>(h2, slotw, out);
}
